// QuantizedBasicBlock_10136122819181
// MI455X (gfx1250) — 8-bit matrix operand order repaired and checked by reading, compiled, not run on hardware
//
#include <hip/hip_runtime.h>
#include <stdint.h>

typedef __attribute__((ext_vector_type(8))) int v8i;

#define NB 32
#define CC 256
#define HH 32
#define WW 32
#define HP 34
#define WP 34

static constexpr int SZP   = NB * HP * WP * CC;
static constexpr int KTOT  = CC * 9;
static constexpr int WPACK = KTOT * CC;

__global__ __launch_bounds__(256) void fill_zero_b128(uint4* __restrict__ p, int n16) {
  int i = blockIdx.x * blockDim.x + threadIdx.x;
  if (i < n16) p[i] = make_uint4(0u, 0u, 0u, 0u);
}

__global__ __launch_bounds__(256) void pack_x_kernel(const int* __restrict__ x,
                                                     const float* __restrict__ z1p,
                                                     int8_t* __restrict__ xp) {
  int t = blockIdx.x * blockDim.x + threadIdx.x;
  int c = t & 255;
  int w = (t >> 8) & 31;
  int h = (t >> 13) & 31;
  int n = t >> 18;
  int z = (int)z1p[0];
  int v = x[((n * CC + c) * HH + h) * WW + w] - z;
  xp[((n * HP + h + 1) * WP + (w + 1)) * CC + c] = (int8_t)v;
}

__global__ __launch_bounds__(256) void pack_w_kernel(const float* __restrict__ w,
                                                     int8_t* __restrict__ wp) {
  int t = blockIdx.x * blockDim.x + threadIdx.x;
  int byte = t & 31;
  int lane = (t >> 5) & 31;
  int nt   = (t >> 10) & 15;
  int kt   = t >> 14;
  int hi   = lane >> 4;
  int kl   = (byte >> 3) * 16 + hi * 8 + (byte & 7);
  int k    = kt * 64 + kl;
  int o    = nt * 16 + (lane & 15);
  int c    = k & 255;
  int sp   = k >> 8;
  int kh   = sp / 3, kw = sp - kh * 3;
  float f  = w[(o * CC + c) * 9 + kh * 3 + kw];
  wp[t] = (int8_t)(int)f;
}

__global__ __launch_bounds__(256, 1) void qconv_gemm_wmma(
    const uint8_t* __restrict__ act,
    const uint8_t* __restrict__ wpk,
    const float*   __restrict__ bias,
    const uint8_t* __restrict__ xsc,
    int8_t*        __restrict__ outMid,
    float*         __restrict__ outF,
    const float* pA, const float* pB, const float* pC,
    const float* pZo,
    const float* pS1, const float* pSo2, const float* pS3, const float* pZ3,
    int isConv2)
{
  const int lane  = threadIdx.x & 31;
  const int waveI = threadIdx.x >> 5;
  const int waveM = waveI >> 2;
  const int waveN = waveI & 3;
  const int r     = lane & 15;
  const int hi    = lane >> 4;

  const int mBase = blockIdx.x * 128 + waveM * 64;
  const int oBase = waveN * 64;

  int M0[4];
  uint32_t aVo[4];
#pragma unroll
  for (int mt = 0; mt < 4; ++mt) {
    int m0  = mBase + mt * 16;
    M0[mt]  = m0;
    int n   = m0 >> 10;
    int oh  = (m0 >> 5) & 31;
    int owb = m0 & 31;
    int rowBase = (n * HP + oh) * WP + owb;
    aVo[mt] = (uint32_t)(((rowBase + r) << 8) + hi * 8);
  }

  uint32_t bVo[4];
#pragma unroll
  for (int nt = 0; nt < 4; ++nt)
    bVo[nt] = (uint32_t)((((waveN * 4 + nt) * 32 + lane) << 5));

  const v8i vzero = {0, 0, 0, 0, 0, 0, 0, 0};
  v8i acc[4][4];
#pragma unroll
  for (int mt = 0; mt < 4; ++mt)
#pragma unroll
    for (int nt = 0; nt < 4; ++nt) acc[mt][nt] = vzero;

#pragma unroll
  for (int kh = 0; kh < 3; ++kh) {
#pragma unroll
    for (int kw = 0; kw < 3; ++kw) {
#pragma unroll
      for (int ct = 0; ct < 4; ++ct) {
        const int ktile = (kh * 3 + kw) * 4 + ct;
        const int aOff  = ((kh * WP + kw) << 8) + ct * 64;
        const int bOff  = ktile << 14;
        v8i a[4], b[4];

#pragma unroll
        for (int mt = 0; mt < 4; ++mt) {
          const uint8_t* ap = act + (size_t)aVo[mt];
          uint2 c0 = *(const uint2*)(ap + (aOff + 0));
          uint2 c1 = *(const uint2*)(ap + (aOff + 16));
          uint2 c2 = *(const uint2*)(ap + (aOff + 32));
          uint2 c3 = *(const uint2*)(ap + (aOff + 48));
          v8i f;
          f[0] = (int)c0.x; f[1] = (int)c0.y;
          f[2] = (int)c1.x; f[3] = (int)c1.y;
          f[4] = (int)c2.x; f[5] = (int)c2.y;
          f[6] = (int)c3.x; f[7] = (int)c3.y;
          a[mt] = f;
        }

#pragma unroll
        for (int nt = 0; nt < 4; ++nt) {
          const uint8_t* bp = wpk + (size_t)bVo[nt];
          uint4 b0 = *(const uint4*)(bp + bOff);
          uint4 b1 = *(const uint4*)(bp + (bOff + 16));
          v8i f;
          f[0] = (int)b0.x; f[1] = (int)b0.y; f[2] = (int)b0.z; f[3] = (int)b0.w;
          f[4] = (int)b1.x; f[5] = (int)b1.y; f[6] = (int)b1.z; f[7] = (int)b1.w;
          b[nt] = f;
        }
        if (ktile < 35)
          __builtin_prefetch(wpk + (size_t)bVo[0] + (bOff + (1 << 14)), 0, 1);

#pragma unroll
        for (int mt = 0; mt < 4; ++mt)
#pragma unroll
          for (int nt = 0; nt < 4; ++nt)
            acc[mt][nt] = __builtin_amdgcn_wmma_i32_16x16x64_iu8(
                true, a[mt], true, b[nt], acc[mt][nt], false, false);
      }
    }
  }

  const float mult = pA[0] * pB[0] / pC[0];
  const float zo   = pZo[0];

  if (!isConv2) {
#pragma unroll
    for (int mt = 0; mt < 4; ++mt) {
      int n   = M0[mt] >> 10;
      int oh  = (M0[mt] >> 5) & 31;
      int owb = M0[mt] & 31;
      uint32_t dVo = (uint32_t)((((n * HP + oh + 1) * WP + owb + 1 + hi * 8) << 8)
                                + oBase + r);
#pragma unroll
      for (int nt = 0; nt < 4; ++nt) {
        float bs = bias[oBase + nt * 16 + r];
#pragma unroll
        for (int i = 0; i < 8; ++i) {
          float accf = (float)acc[mt][nt][i] + bs;
          float q = rintf(accf * mult) + zo;
          q = fminf(fmaxf(q, 0.0f), 255.0f);
          outMid[(size_t)dVo + (i << 8) + nt * 16] = (int8_t)(int)(q - zo);
        }
      }
    }
  } else {
    const float r13 = pS1[0]  / pS3[0];
    const float r23 = pSo2[0] / pS3[0];
    const float z3  = pZ3[0];
#pragma unroll
    for (int mt = 0; mt < 4; ++mt) {
      int n   = M0[mt] >> 10;
      int oh  = (M0[mt] >> 5) & 31;
      int owb = M0[mt] & 31;
      uint32_t sVo = (uint32_t)((((n * HP + oh + 1) * WP + owb + 1 + hi * 8) << 8)
                                + oBase + r);
      uint32_t oVo = (uint32_t)(((((n * CC + oBase + r) * HH + oh) * WW)
                                 + owb + hi * 8) * 4);
#pragma unroll
      for (int nt = 0; nt < 4; ++nt) {
        float bs = bias[oBase + nt * 16 + r];
#pragma unroll
        for (int i = 0; i < 8; ++i) {
          float accf = (float)acc[mt][nt][i] + bs;
          float q2 = rintf(accf * mult) + zo;
          q2 = fminf(fmaxf(q2, 0.0f), 255.0f);
          float q2m = q2 - zo;
          float xv = (float)(int8_t)xsc[(size_t)sVo + (i << 8) + nt * 16];
          float outv = rintf(r13 * xv) + rintf(r23 * q2m) + z3;
          outv = fminf(fmaxf(outv, 0.0f), 255.0f);
          *(float*)((uint8_t*)outF + (size_t)oVo
                    + (size_t)(nt * 16 * HH * WW * 4) + (i * 4)) = outv;
        }
      }
    }
  }
}

extern "C" void kernel_launch(void* const* d_in, const int* in_sizes, int n_in,
                              void* d_out, int out_size, void* d_ws, size_t ws_size,
                              hipStream_t stream) {
  const int*   x      = (const int*)d_in[0];
  const float* w1     = (const float*)d_in[1];
  const float* b1     = (const float*)d_in[2];
  const float* w2     = (const float*)d_in[3];
  const float* b2     = (const float*)d_in[4];
  const float* s1     = (const float*)d_in[5];
  const float* z1     = (const float*)d_in[6];
  const float* sw1    = (const float*)d_in[7];
  const float* s_mid  = (const float*)d_in[8];
  const float* z_mid  = (const float*)d_in[9];
  const float* sw2    = (const float*)d_in[10];
  const float* s_out2 = (const float*)d_in[11];
  const float* z_out2 = (const float*)d_in[12];
  const float* s3     = (const float*)d_in[13];
  const float* z3     = (const float*)d_in[14];

  uint8_t* ws     = (uint8_t*)d_ws;
  uint8_t* xPad   = ws;
  uint8_t* midPad = ws + SZP;
  uint8_t* wp1    = ws + 2 * (size_t)SZP;
  uint8_t* wp2    = ws + 2 * (size_t)SZP + WPACK;

  int n16 = (2 * SZP) / 16;
  fill_zero_b128<<<(n16 + 255) / 256, 256, 0, stream>>>((uint4*)ws, n16);

  pack_x_kernel<<<(NB * HH * WW * CC) / 256, 256, 0, stream>>>(x, z1, (int8_t*)xPad);
  pack_w_kernel<<<WPACK / 256, 256, 0, stream>>>(w1, (int8_t*)wp1);
  pack_w_kernel<<<WPACK / 256, 256, 0, stream>>>(w2, (int8_t*)wp2);

  const int mBlocks = (NB * HH * WW) / 128;
  qconv_gemm_wmma<<<mBlocks, 256, 0, stream>>>(
      xPad, wp1, b1, nullptr, (int8_t*)midPad, nullptr,
      s1, sw1, s_mid, z_mid, s1, s_out2, s3, z3, 0);

  qconv_gemm_wmma<<<mBlocks, 256, 0, stream>>>(
      midPad, wp2, b2, xPad, nullptr, (float*)d_out,
      s_mid, sw2, s_out2, z_out2, s1, s_out2, s3, z3, 1);
}
